// Inception_Attention_Block_9586367005343
// MI455X (gfx1250) — compile-verified, matrix operand order checked by reading, not run on hardware
//
#include <hip/hip_runtime.h>

typedef _Float16 half_t;
typedef __attribute__((ext_vector_type(16))) _Float16 v16h;
typedef __attribute__((ext_vector_type(8)))  _Float16 v8h;
typedef __attribute__((ext_vector_type(8)))  float    v8f;

#define KS 40   // padded K-stride in halfs (80B = 20 banks -> conflict-free row access)

// ---------------------------------------------------------------------------
// Fused q/k projection + scores + softmax, one workgroup per batch index n.
// attn[n, :, :] written exactly once. Per-lane online softmax in the tile
// loop (no cross-lane ops); single 16-lane (m,s) merge per 16-row band.
// ---------------------------------------------------------------------------
template<int P, int W, int NT>
__global__ __launch_bounds__(256) void attn_kernel(
    const float* __restrict__ x,
    const float* __restrict__ wq, const float* __restrict__ bq,
    const float* __restrict__ wk, const float* __restrict__ bk,
    float* __restrict__ attn)
{
    constexpr int WPAD = NT * 16;
    constexpr int C = 256, L = 2000;

    __shared__ float z_lds[W * P];
    __shared__ float wq_lds[P * P], wk_lds[P * P];
    __shared__ float bq_lds[P], bk_lds[P];
    __shared__ __align__(16) half_t q_lds[WPAD * KS];
    __shared__ __align__(16) half_t k_lds[WPAD * KS];

    const int n   = blockIdx.x;
    const int b   = n >> 8;     // C == 256
    const int c   = n & 255;
    const int tid = threadIdx.x;

    // --- stage weights + this batch's z panel (z[l] = x[b, l, c]) ---
    for (int i = tid; i < P * P; i += 256) { wq_lds[i] = wq[i]; wk_lds[i] = wk[i]; }
    if (tid < P) { bq_lds[tid] = bq[tid]; bk_lds[tid] = bk[tid]; }
    const float* xb = x + (size_t)b * L * C + c;
    for (int l = tid; l < W * P; l += 256) z_lds[l] = xb[(size_t)l * C];
    for (int i = tid; i < WPAD * KS; i += 256) { q_lds[i] = (half_t)0.f; k_lds[i] = (half_t)0.f; }
    __syncthreads();

    // --- q = z wq^T + bq (scaled by 1/sqrt(P)), k = z wk^T + bk, stored fp16 ---
    const float qscale = rsqrtf((float)P);
    for (int idx = tid; idx < W * P; idx += 256) {
        const int i = idx / P, j = idx % P;
        float aq = bq_lds[j], ak = bk_lds[j];
        const float* zr  = &z_lds[i * P];
        const float* wqr = &wq_lds[j * P];
        const float* wkr = &wk_lds[j * P];
        #pragma unroll
        for (int t = 0; t < P; ++t) { const float zv = zr[t]; aq += zv * wqr[t]; ak += zv * wkr[t]; }
        q_lds[i * KS + j] = (half_t)(aq * qscale);
        k_lds[i * KS + j] = (half_t)ak;
    }
    __syncthreads();

    // --- scores + softmax, 16-row band per wave iteration ---
    const int wv     = tid >> 5;
    const int lane   = tid & 31;
    const int laneLo = lane & 15;
    const int hi     = lane >> 4;     // A/B fragment K-group select (ISA 16-bit layout)

    float* outn = attn + (size_t)n * W * W;

    for (int tr = wv; tr < NT; tr += 8) {
        // A fragment: q rows tr*16+laneLo; halves 0..7 -> K = hi*8..hi*8+7,
        // halves 8..15 -> K = 16+hi*8.. (two contiguous 16B LDS loads)
        const int arow = tr * 16 + laneLo;
        const v8h a0 = *(const v8h*)&q_lds[arow * KS + hi * 8];
        const v8h a1 = *(const v8h*)&q_lds[arow * KS + 16 + hi * 8];
        const v16h A = __builtin_shufflevector(a0, a1,
            0, 1, 2, 3, 4, 5, 6, 7, 8, 9, 10, 11, 12, 13, 14, 15);

        // per-lane running (max, sum) over this lane's column subset
        float m[8], s[8];
        #pragma unroll
        for (int r = 0; r < 8; ++r) { m[r] = -3.0e30f; s[r] = 0.f; }

        // sweep 1: per-lane online softmax stats (no cross-lane traffic)
        for (int tc = 0; tc < NT; ++tc) {
            const int brow = tc * 16 + laneLo;
            const v8h b0 = *(const v8h*)&k_lds[brow * KS + hi * 8];
            const v8h b1 = *(const v8h*)&k_lds[brow * KS + 16 + hi * 8];
            const v16h Bm = __builtin_shufflevector(b0, b1,
                0, 1, 2, 3, 4, 5, 6, 7, 8, 9, 10, 11, 12, 13, 14, 15);
            v8f Cacc = {};
            Cacc = __builtin_amdgcn_wmma_f32_16x16x32_f16(false, A, false, Bm,
                                                          (short)0, Cacc, false, false);
            const bool colOK = (tc * 16 + laneLo) < W;
            #pragma unroll
            for (int r = 0; r < 8; ++r) {
                const float v = colOK ? Cacc[r] : -__builtin_inff();
                const float d = v - m[r];
                const float e = __expf(-__builtin_fabsf(d));   // one exp per reg per tile
                if (d > 0.f) { s[r] = s[r] * e + 1.f; m[r] = v; }
                else         { s[r] += e; }
            }
        }

        // one (m,s) merge across the 16-lane half-group, then invert s
        #pragma unroll
        for (int r = 0; r < 8; ++r) {
            #pragma unroll
            for (int msk = 1; msk < 16; msk <<= 1) {
                const float mo = __shfl_xor(m[r], msk, 32);
                const float so = __shfl_xor(s[r], msk, 32);
                const float mn = fmaxf(m[r], mo);
                s[r] = s[r] * __expf(m[r] - mn) + so * __expf(mo - mn);
                m[r] = mn;
            }
            s[r] = 1.f / s[r];
        }

        // sweep 2: recompute tile, write normalized attn (coalesced 64B rows)
        for (int tc = 0; tc < NT; ++tc) {
            const int col = tc * 16 + laneLo;
            const v8h b0 = *(const v8h*)&k_lds[col * KS + hi * 8];
            const v8h b1 = *(const v8h*)&k_lds[col * KS + 16 + hi * 8];
            const v16h Bm = __builtin_shufflevector(b0, b1,
                0, 1, 2, 3, 4, 5, 6, 7, 8, 9, 10, 11, 12, 13, 14, 15);
            v8f Cacc = {};
            Cacc = __builtin_amdgcn_wmma_f32_16x16x32_f16(false, A, false, Bm,
                                                          (short)0, Cacc, false, false);
            if (col < W) {
                #pragma unroll
                for (int r = 0; r < 8; ++r) {
                    const int row = tr * 16 + hi * 8 + r;
                    if (row < W)
                        outn[(size_t)row * W + col] = __expf(Cacc[r] - m[r]) * s[r];
                }
            }
        }
    }
}

// ---------------------------------------------------------------------------
// res[b,l,c]: closed-form interpolation.
//   p=10: frac==0.5 -> 0.5*(z2_10[20l+9] + z2_10[20l+10]); both cols share z-row l/10
//   p=20: frac==0   -> z2_20[5l+2]; z-row l/20
// One block per (b, l-group of 20); threads = c (fully coalesced x reads/res writes).
// ---------------------------------------------------------------------------
__global__ __launch_bounds__(256) void res_kernel(
    const float* __restrict__ x,
    const float* __restrict__ wl10, const float* __restrict__ bl10,
    const float* __restrict__ wl20, const float* __restrict__ bl20,
    float* __restrict__ res)
{
    constexpr int C = 256, L = 2000;
    const int c   = threadIdx.x;
    const int blk = blockIdx.x;          // B * (L/20) = 800
    const int b   = blk / 100;
    const int a   = blk % 100;           // l in [20a, 20a+20)

    const float* xb = x + ((size_t)b * L + a * 20) * C + c;
    float xr[20];
    #pragma unroll
    for (int j = 0; j < 20; ++j) xr[j] = xb[(size_t)j * C];

    float* outb = res + ((size_t)b * L + a * 20) * C + c;

    for (int t = 0; t < 20; ++t) {
        const int t10    = t % 10;
        const int dec    = (t / 10) * 10;
        const int col10a = 20 * t10 + 9;
        const int col10b = col10a + 1;
        const int col20  = 5 * t + 2;

        float e10a = bl10[col10a], e10b = bl10[col10b];
        const float* w10a = wl10 + col10a * 10;
        const float* w10b = wl10 + col10b * 10;
        #pragma unroll
        for (int j = 0; j < 10; ++j) {
            const float zv = xr[dec + j];
            e10a += zv * w10a[j];
            e10b += zv * w10b[j];
        }
        float e20 = bl20[col20];
        const float* w20 = wl20 + col20 * 20;
        #pragma unroll
        for (int j = 0; j < 20; ++j) e20 += xr[j] * w20[j];

        outb[(size_t)t * C] = 0.25f * (e10a + e10b) + 0.5f * e20;
    }
}

extern "C" void kernel_launch(void* const* d_in, const int* in_sizes, int n_in,
                              void* d_out, int out_size, void* d_ws, size_t ws_size,
                              hipStream_t stream) {
    const float* x    = (const float*)d_in[0];
    const float* wq10 = (const float*)d_in[1];
    const float* bq10 = (const float*)d_in[2];
    const float* wk10 = (const float*)d_in[3];
    const float* bk10 = (const float*)d_in[4];
    const float* wl10 = (const float*)d_in[5];
    const float* bl10 = (const float*)d_in[6];
    const float* wq20 = (const float*)d_in[7];
    const float* bq20 = (const float*)d_in[8];
    const float* wk20 = (const float*)d_in[9];
    const float* bk20 = (const float*)d_in[10];
    const float* wl20 = (const float*)d_in[11];
    const float* bl20 = (const float*)d_in[12];

    float* res    = (float*)d_out;                       // 8*2000*256
    float* attn10 = res    + (size_t)8 * 2000 * 256;     // 2048*200*200
    float* attn20 = attn10 + (size_t)2048 * 200 * 200;   // 2048*100*100

    res_kernel<<<800, 256, 0, stream>>>(x, wl10, bl10, wl20, bl20, res);
    attn_kernel<10, 200, 13><<<2048, 256, 0, stream>>>(x, wq10, bq10, wk10, bk10, attn10);
    attn_kernel<20, 100, 7 ><<<2048, 256, 0, stream>>>(x, wq20, bq20, wk20, bk20, attn20);
}
